// PathwayGroupProjection_29678224015958
// MI455X (gfx1250) — compile-verified, matrix operand order checked by reading, not run on hardware
//
#include <hip/hip_runtime.h>
#include <hip/hip_bf16.h>
#include <math.h>

typedef __attribute__((ext_vector_type(16))) __bf16       v16bf;
typedef __attribute__((ext_vector_type(8)))  float        v8f;
typedef __attribute__((ext_vector_type(8)))  unsigned int v8u;
typedef __attribute__((ext_vector_type(4)))  unsigned int v4u;

constexpr int B = 256, G = 256, S = 128, D = 256;
constexpr float EPS = 1e-5f;
constexpr int NTHREADS = 512;                 // 16 wave32 waves per WG

// LDS strides in bf16 elements. Strides are multiples of 8 elems (16 B) so the
// two 16-byte halves of every WMMA fragment are b128-alignable, and stride/4
// (dwords) == 4 (mod 64) so the 16 lanes of a half sweep all 64 LDS banks.
constexpr int XA_STRIDE = S + 8;              // 136 elems = 272 B
constexpr int W1_STRIDE = S + 8;              // 136
constexpr int HA_STRIDE = D + 8;              // 264 elems = 528 B
constexpr int W2_STRIDE = D + 8;              // 264

constexpr int XA_BYTES = B * XA_STRIDE * 2;   // 69632
constexpr int W1_BYTES = D * W1_STRIDE * 2;   // 69632
constexpr int HA_BYTES = B * HA_STRIDE * 2;   // 135168
constexpr int W2_BYTES = D * W2_STRIDE * 2;   // 135168
constexpr int REGION0  = XA_BYTES + W1_BYTES; // 139264 >= W2_BYTES (W2T overlays x/W1)
constexpr size_t LDS_BYTES = (size_t)REGION0 + HA_BYTES;  // 274432 <= 320KB WGP LDS

// fp32 -> bf16 (RNE), as raw 16-bit pattern
__device__ __forceinline__ unsigned f2bfu(float f) {
  unsigned u = __builtin_bit_cast(unsigned, f);
  u += 0x7FFFu + ((u >> 16) & 1u);
  return u >> 16;
}
__device__ __forceinline__ __bf16 f2bf(float f) {
  unsigned short h = (unsigned short)f2bfu(f);
  return __builtin_bit_cast(__bf16, h);
}

// 16x32 bf16 WMMA fragment (A layout; B identical with lane = column).
// CDNA5 16-bit layout: lane half h, VGPR0..3 = K h*8+0..7, VGPR4..7 = K 16+h*8+0..7
// -> two contiguous 16-byte runs per lane -> two ds_load_b128.
__device__ __forceinline__ v16bf load_frag(const __bf16* base, int stride, int lane) {
  const int m    = lane & 15;
  const int half = lane >> 4;
  const __bf16* p = base + m * stride + half * 8;
  const v4u lo = *reinterpret_cast<const v4u*>(p);        // K = half*8 + 0..7
  const v4u hi = *reinterpret_cast<const v4u*>(p + 16);   // K = 16 + half*8 + 0..7
  const v8u r  = __builtin_shufflevector(lo, hi, 0, 1, 2, 3, 4, 5, 6, 7);
  return __builtin_bit_cast(v16bf, r);
}

__global__ __launch_bounds__(NTHREADS, 1)
void pathway_group_kernel(const float* __restrict__ x,
                          const float* __restrict__ W1,
                          const float* __restrict__ b1,
                          const float* __restrict__ ln_g,
                          const float* __restrict__ ln_b,
                          const float* __restrict__ W2,
                          const float* __restrict__ b2,
                          float* __restrict__ out) {
  extern __shared__ char smem[];
  __bf16* xA  = reinterpret_cast<__bf16*>(smem);
  __bf16* w1T = reinterpret_cast<__bf16*>(smem + XA_BYTES);
  __bf16* w2T = reinterpret_cast<__bf16*>(smem);            // overlays xA/w1T in phase 2
  __bf16* hA  = reinterpret_cast<__bf16*>(smem + REGION0);

  const int g    = blockIdx.x;
  const int tid  = threadIdx.x;
  const int lane = tid & 31;
  const int wave = tid >> 5;
  const int col  = lane & 15;       // output column within 16-wide tile
  const int half = lane >> 4;       // C/D: lanes 16-31 hold rows M=8..15
  const int m0   = wave * 16;       // this wave owns batch rows m0..m0+15

  const float* W1g = W1 + (size_t)g * S * D;
  const float* W2g = W2 + (size_t)g * D * D;

  // Speculative prefetch of W2 (needed in phase 2) into GL2 while we do GEMM1
  for (int i = tid * 16; i < D * D; i += NTHREADS * 16)
    __builtin_prefetch(W2g + i, 0, 1);

  // ---------- stage x tile: 2 elems/iter -> b64 global read + packed b32 LDS store ----------
  for (int i = tid; i < (B * S) / 2; i += NTHREADS) {
    const int b = i >> 6, s = (i & 63) * 2;                 // S/2 == 64
    const float* px = x + (size_t)b * (G * S) + (size_t)g * S + s;
    const unsigned packed = f2bfu(px[0]) | (f2bfu(px[1]) << 16);
    *reinterpret_cast<unsigned*>(xA + b * XA_STRIDE + s) = packed;
  }
  // ---------- stage W1^T (coalesced global reads over n; transposed b16 LDS writes) ----------
  for (int i = tid; i < S * D; i += NTHREADS) {
    const int k = i >> 8, n = i & (D - 1);
    w1T[n * W1_STRIDE + k] = f2bf(W1g[k * D + n]);
  }
  __syncthreads();

  // ---------- GEMM1: h = x_tile @ W1  (M=16/wave, K=128, N=256) ----------
  v16bf a1[4];
#pragma unroll
  for (int kc = 0; kc < 4; ++kc)
    a1[kc] = load_frag(xA + m0 * XA_STRIDE + kc * 32, XA_STRIDE, lane);

  const v8f vzero = {0.f, 0.f, 0.f, 0.f, 0.f, 0.f, 0.f, 0.f};
  v8f acc[16];
#pragma unroll
  for (int t = 0; t < 16; ++t) acc[t] = vzero;

#pragma unroll
  for (int t = 0; t < 16; ++t) {
#pragma unroll
    for (int kc = 0; kc < 4; ++kc) {
      const v16bf bf = load_frag(w1T + (t * 16) * W1_STRIDE + kc * 32, W1_STRIDE, lane);
      acc[t] = __builtin_amdgcn_wmma_f32_16x16x32_bf16(
          false, a1[kc], false, bf, (short)0, acc[t], false, false);
    }
  }
  __syncthreads();  // all xA/w1T reads done -> region0 may be overwritten by W2^T

  // ---------- bias + LayerNorm statistics (rows live inside this wave) ----------
  float s1[8], s2[8];
#pragma unroll
  for (int vg = 0; vg < 8; ++vg) { s1[vg] = 0.f; s2[vg] = 0.f; }
#pragma unroll
  for (int t = 0; t < 16; ++t) {
    const float bias = b1[(size_t)g * D + t * 16 + col];
#pragma unroll
    for (int vg = 0; vg < 8; ++vg) {
      const float v = acc[t][vg] + bias;
      acc[t][vg] = v;
      s1[vg] += v;
      s2[vg] += v * v;
    }
  }
  // butterfly over the 16 lanes that share each row (xor<16 stays in lane-half)
#pragma unroll
  for (int vg = 0; vg < 8; ++vg) {
#pragma unroll
    for (int off = 1; off < 16; off <<= 1) {
      s1[vg] += __shfl_xor(s1[vg], off, 32);
      s2[vg] += __shfl_xor(s2[vg], off, 32);
    }
  }
  float mu[8], rstd[8];
#pragma unroll
  for (int vg = 0; vg < 8; ++vg) {
    mu[vg] = s1[vg] * (1.f / D);
    const float var = s2[vg] * (1.f / D) - mu[vg] * mu[vg];
    rstd[vg] = rsqrtf(var + EPS);
  }

  // ---------- LN affine + exact GELU -> hA (bf16); co-stage W2^T ----------
#pragma unroll
  for (int t = 0; t < 16; ++t) {
    const int n = t * 16 + col;
    const float gamma = ln_g[(size_t)g * D + n];
    const float beta  = ln_b[(size_t)g * D + n];
#pragma unroll
    for (int vg = 0; vg < 8; ++vg) {
      float v = (acc[t][vg] - mu[vg]) * rstd[vg] * gamma + beta;
      v = 0.5f * v * (1.0f + erff(v * 0.70710678118654752440f));   // exact GELU
      hA[(m0 + vg + half * 8) * HA_STRIDE + n] = f2bf(v);
    }
  }
  for (int i = tid; i < D * D; i += NTHREADS) {
    const int k = i >> 8, n = i & (D - 1);
    w2T[n * W2_STRIDE + k] = f2bf(W2g[k * D + n]);
  }
  __syncthreads();

  // ---------- GEMM2: out = gelu(h) @ W2 + b2  (K=256, N=256) ----------
  v16bf a2[8];
#pragma unroll
  for (int kc = 0; kc < 8; ++kc)
    a2[kc] = load_frag(hA + m0 * HA_STRIDE + kc * 32, HA_STRIDE, lane);

#pragma unroll
  for (int t = 0; t < 16; ++t) acc[t] = vzero;
#pragma unroll
  for (int t = 0; t < 16; ++t) {
#pragma unroll
    for (int kc = 0; kc < 8; ++kc) {
      const v16bf bf = load_frag(w2T + (t * 16) * W2_STRIDE + kc * 32, W2_STRIDE, lane);
      acc[t] = __builtin_amdgcn_wmma_f32_16x16x32_bf16(
          false, a2[kc], false, bf, (short)0, acc[t], false, false);
    }
  }

  // ---------- epilogue: out[row][g][n] = acc + b2[g][n] ----------
#pragma unroll
  for (int t = 0; t < 16; ++t) {
    const int n = t * 16 + col;
    const float bias = b2[(size_t)g * D + n];
#pragma unroll
    for (int vg = 0; vg < 8; ++vg) {
      const int row = m0 + vg + half * 8;
      out[((size_t)row * G + g) * D + n] = acc[t][vg] + bias;
    }
  }
}

extern "C" void kernel_launch(void* const* d_in, const int* in_sizes, int n_in,
                              void* d_out, int out_size, void* d_ws, size_t ws_size,
                              hipStream_t stream) {
  (void)in_sizes; (void)n_in; (void)out_size; (void)d_ws; (void)ws_size;
  const float* x    = (const float*)d_in[0];
  const float* W1   = (const float*)d_in[1];
  const float* b1   = (const float*)d_in[2];
  const float* ln_g = (const float*)d_in[3];
  const float* ln_b = (const float*)d_in[4];
  const float* W2   = (const float*)d_in[5];
  const float* b2   = (const float*)d_in[6];
  float* out = (float*)d_out;

  dim3 grid(G);            // one workgroup per group: weights read exactly once
  dim3 block(NTHREADS);    // 16 waves; each owns 16 batch rows
  pathway_group_kernel<<<grid, block, LDS_BYTES, stream>>>(
      x, W1, b1, ln_g, ln_b, W2, b2, out);
}
